// IPMPDecoder_18897856103197
// MI455X (gfx1250) — matrix operand order repaired and checked by reading, not compiled, not run on hardware
//
#include <hip/hip_runtime.h>
#include <hip/hip_bf16.h>
#include <math.h>

// ---------------- problem constants ----------------
#define NN   4096
#define KNB  30
#define EE   (NN * KNB)      // 122880 edges
#define CS_  256
#define CZ_  128
#define CH_  128
#define HET_ 384
#define EFD  416             // 25*16 + 16 edge feature dim

typedef __attribute__((ext_vector_type(16))) __bf16 bf16x16;
typedef __attribute__((ext_vector_type(8)))  __bf16 bf16x8;
typedef __attribute__((ext_vector_type(8)))  float  f32x8;
typedef unsigned int u32x4 __attribute__((ext_vector_type(4)));
typedef int          i32x4 __attribute__((ext_vector_type(4)));
typedef int          i32x8 __attribute__((ext_vector_type(8)));

#define TDM_OK __has_builtin(__builtin_amdgcn_tensor_load_to_lds)

// =====================================================================
// TDM: DMA one B tile (128 rows x 32 k of bf16, row stride Kd elems)
// from pre-transposed weights Wt[N][K] into LDS at ldsOff.
// tensor_dims = remaining K/N so hardware OOB-clamps tails to zero.
// =====================================================================
__device__ __forceinline__ void tdm_load_b_tile(
    const __bf16* Wt, unsigned ldsOff, int Kd, int Nd, int tileN, int kk)
{
#if TDM_OK
  unsigned long long ga = (unsigned long long)(Wt + (size_t)tileN * Kd + kk);
  u32x4 g0;
  g0[0] = 1u;                                                  // count=1 (valid D#)
  g0[1] = ldsOff;                                              // lds_addr [63:32]
  g0[2] = (unsigned)ga;                                        // global_addr lo
  g0[3] = ((unsigned)(ga >> 32) & 0x01FFFFFFu) | (2u << 30);   // addr[56:32] | type=2
  unsigned td0 = (unsigned)(Kd - kk);                          // tensor_dim0 (elems)
  unsigned td1 = (unsigned)(Nd - tileN);                       // tensor_dim1
  i32x8 g1;
  g1[0] = (int)(1u << 16);                                     // data_size = 2 bytes
  g1[1] = (int)((td0 & 0xFFFFu) << 16);                        // tensor_dim0 [79:48]
  g1[2] = (int)((td0 >> 16) | ((td1 & 0xFFFFu) << 16));        // tensor_dim1 [111:80]
  g1[3] = (int)((td1 >> 16) | (32u << 16));                    // tile_dim0 = 32
  g1[4] = (int)128u;                                           // tile_dim1 = 128
  g1[5] = (int)(unsigned)Kd;                                   // tensor_dim0_stride lo
  g1[6] = 0;
  g1[7] = 0;
  i32x4 z4 = {0, 0, 0, 0};
#if __clang_major__ >= 23
  i32x8 z8 = {0, 0, 0, 0, 0, 0, 0, 0};
  __builtin_amdgcn_tensor_load_to_lds(g0, g1, z4, z4, z8, 0);
#else
  __builtin_amdgcn_tensor_load_to_lds(g0, g1, z4, z4, 0);
#endif
  __builtin_amdgcn_s_wait_tensorcnt(0);
#endif
}

// =====================================================================
// Weight pre-transpose: W[K][N] f32 -> Wt[N][K] bf16 (one-time, L2 resident)
// =====================================================================
__global__ __launch_bounds__(256) void wtrans_kernel(
    const float* __restrict__ W, __bf16* __restrict__ Wt, int Kd, int Nd)
{
  int idx = blockIdx.x * blockDim.x + threadIdx.x;
  if (idx >= Kd * Nd) return;
  int n = idx / Kd, k = idx % Kd;
  Wt[idx] = (__bf16)W[(size_t)k * Nd + n];
}

// =====================================================================
// GEMM:  C[M,Nd] (+)= act( A[M,Kd] @ W[Kd,Nd] + bias )
//   Wt: bf16 pre-transposed weights [Nd][Kd]
//   rowIdx: optional gather of A rows (s[centers] / s[neigh])
//   beta=1 -> accumulate into C; act=1 -> ReLU
// 128x128 tile, 256 threads = 8 wave32, wave owns a 16-row strip.
// BOUNDS=false: fully exact tiles, no guards (the common case here).
// =====================================================================
template <bool BOUNDS>
__global__ __launch_bounds__(256) void gemm_bf16_wmma(
    const float* __restrict__ A, const int* __restrict__ rowIdx,
    const __bf16* __restrict__ Wt, const float* __restrict__ bias,
    float* __restrict__ C, int M, int Nd, int Kd, int beta, int act)
{
  __shared__ __align__(32) __bf16 lA[128][32];   // [m][k]
  __shared__ __align__(32) __bf16 lB[128][32];   // [n][k]

  const int tid  = threadIdx.x;
  const int wave = tid >> 5;
  const int lane = tid & 31;
  const int hl   = lane >> 4;
  const int l16  = lane & 15;
  const int tileM = blockIdx.x * 128;
  const int tileN = blockIdx.y * 128;
  const unsigned ldsOffB = (unsigned)(size_t)(void*)&lB[0][0];

  // ---- hoisted A row gather (independent of kk) ----
  const float* aptr[4];
#pragma unroll
  for (int rr = 0; rr < 4; ++rr) {
    int grow = tileM + rr * 32 + (tid >> 3);
    bool ok = !BOUNDS || (grow < M);
    int ar = 0;
    if (ok) ar = rowIdx ? rowIdx[grow] : grow;
    aptr[rr] = ok ? (A + (size_t)ar * Kd + (tid & 7) * 4) : nullptr;
  }

  f32x8 acc[8] = {};

  for (int kk = 0; kk < Kd; kk += 32) {
    // ---- stage B tile ----
#if TDM_OK
    if (wave == 0) tdm_load_b_tile(Wt, ldsOffB, Kd, Nd, tileN, kk);
#else
    {
      int n = tid >> 1, seg = (tid & 1) * 16;
      __bf16* dst = &lB[n][seg];
      const __bf16* src = Wt + (size_t)(tileN + n) * Kd + kk + seg;
      if (!BOUNDS) {
        *(uint4*)dst       = *(const uint4*)src;
        *(uint4*)(dst + 8) = *(const uint4*)(src + 8);
      } else {
        bool nok = (tileN + n) < Nd;
#pragma unroll
        for (int i = 0; i < 16; ++i) {
          __bf16 v = (__bf16)0.f;
          if (nok && (kk + seg + i) < Kd) v = src[i];
          dst[i] = v;
        }
      }
    }
#endif
    // ---- stage A tile (f32 -> bf16) ----
#pragma unroll
    for (int rr = 0; rr < 4; ++rr) {
      int row = rr * 32 + (tid >> 3);
      int k4  = (tid & 7) * 4;
      float v0 = 0.f, v1 = 0.f, v2 = 0.f, v3 = 0.f;
      if (!BOUNDS || aptr[rr]) {
        const float* p = aptr[rr] + kk;
        if (!BOUNDS || (kk + k4 + 4 <= Kd)) {
          float4 f = *(const float4*)p;
          v0 = f.x; v1 = f.y; v2 = f.z; v3 = f.w;
        } else {
          if (kk + k4 + 0 < Kd) v0 = p[0];
          if (kk + k4 + 1 < Kd) v1 = p[1];
          if (kk + k4 + 2 < Kd) v2 = p[2];
          if (kk + k4 + 3 < Kd) v3 = p[3];
        }
      }
      lA[row][k4 + 0] = (__bf16)v0; lA[row][k4 + 1] = (__bf16)v1;
      lA[row][k4 + 2] = (__bf16)v2; lA[row][k4 + 3] = (__bf16)v3;
    }
    // prefetch next A K-block (global_prefetch_b8)
    if (kk + 32 < Kd && (tid & 7) == 0 && (!BOUNDS || aptr[0]))
      __builtin_prefetch(aptr[0] + kk + 32, 0, 0);

    __syncthreads();

    // ---- A fragment: lane -> row wave*16+l16, K run hl*16 .. hl*16+15 ----
    int arow = wave * 16 + l16;
    bf16x16 afrag = *(const bf16x16*)&lA[arow][hl * 16];

#pragma unroll
    for (int ct = 0; ct < 8; ++ct) {
      bf16x16 bfrag = *(const bf16x16*)&lB[ct * 16 + l16][hl * 16];
      acc[ct] = __builtin_amdgcn_wmma_f32_16x16x32_bf16(
          false, afrag, false, bfrag, (short)0, acc[ct], false, false);
    }
    __syncthreads();
  }

  // ---- epilogue: VGPR r -> M = wave*16 + hl*8 + r, N = ct*16 + l16 ----
#pragma unroll
  for (int ct = 0; ct < 8; ++ct) {
    int gcol = tileN + ct * 16 + l16;
#pragma unroll
    for (int r = 0; r < 8; ++r) {
      int grow = tileM + wave * 16 + hl * 8 + r;
      if (!BOUNDS || (grow < M && gcol < Nd)) {
        float v = acc[ct][r];
        if (bias) v += bias[gcol];
        if (beta) v += C[(size_t)grow * Nd + gcol];
        if (act)  v = fmaxf(v, 0.f);
        C[(size_t)grow * Nd + gcol] = v;
      }
    }
  }
}

// =====================================================================
// LayerNorm, one wave32 per row.  mode 0: y = LN(x)
//   mode 1: y = LN(x + upd*nm)      (nm = mask ? 0 : 1)
//   mode 2: y = LN(x + upd) * nm
// =====================================================================
__global__ __launch_bounds__(256) void ln_kernel(
    const float* __restrict__ x, const float* __restrict__ upd,
    const float* __restrict__ g, const float* __restrict__ b,
    const unsigned char* __restrict__ mask, int mode,
    float* __restrict__ out, int M, int D)
{
  int row = blockIdx.x * 8 + (threadIdx.x >> 5);
  if (row >= M) return;
  int lane = threadIdx.x & 31;
  int per  = D >> 5;
  float nm = 1.f;
  if (mask) nm = mask[row] ? 0.f : 1.f;

  float vals[8];
  float sum = 0.f;
  for (int i = 0; i < per; ++i) {
    int c = i * 32 + lane;
    float v = x[(size_t)row * D + c];
    if (upd) {
      float u = upd[(size_t)row * D + c];
      v += (mode == 1) ? u * nm : u;
    }
    vals[i] = v; sum += v;
  }
  for (int o = 16; o; o >>= 1) sum += __shfl_xor(sum, o, 32);
  float mean = sum / (float)D;
  float s2 = 0.f;
  for (int i = 0; i < per; ++i) { float d = vals[i] - mean; s2 += d * d; }
  for (int o = 16; o; o >>= 1) s2 += __shfl_xor(s2, o, 32);
  float rstd = rsqrtf(s2 / (float)D + 1e-5f);
  for (int i = 0; i < per; ++i) {
    int c = i * 32 + lane;
    float y = (vals[i] - mean) * rstd * g[c] + b[c];
    if (mode == 2) y *= nm;
    out[(size_t)row * D + c] = y;
  }
}

// =====================================================================
// Geometry prep: bb5 (add virtual CB), masked coords, squared norms
// =====================================================================
__global__ __launch_bounds__(256) void prep_kernel(
    const float* __restrict__ x_ca, const float* __restrict__ bb,
    const unsigned char* __restrict__ mask,
    float* __restrict__ bb5, float* __restrict__ xm, float* __restrict__ sq)
{
  int n = blockIdx.x * blockDim.x + threadIdx.x;
  if (n >= NN) return;
  const float* p = bb + (size_t)n * 12;
  float nn0 = p[0], nn1 = p[1], nn2 = p[2];
  float ca0 = p[3], ca1 = p[4], ca2 = p[5];
  float cc0 = p[6], cc1 = p[7], cc2 = p[8];
  float bv0 = ca0 - nn0, bv1 = ca1 - nn1, bv2 = ca2 - nn2;
  float cv0 = cc0 - ca0, cv1 = cc1 - ca1, cv2 = cc2 - ca2;
  float av0 = bv1 * cv2 - bv2 * cv1;
  float av1 = bv2 * cv0 - bv0 * cv2;
  float av2 = bv0 * cv1 - bv1 * cv0;
  float cb0 = -0.58273431f * av0 + 0.56802827f * bv0 - 0.54067466f * cv0 + ca0;
  float cb1 = -0.58273431f * av1 + 0.56802827f * bv1 - 0.54067466f * cv1 + ca1;
  float cb2 = -0.58273431f * av2 + 0.56802827f * bv2 - 0.54067466f * cv2 + ca2;
  float* q = bb5 + (size_t)n * 15;
  for (int i = 0; i < 12; ++i) q[i] = p[i];
  q[12] = cb0; q[13] = cb1; q[14] = cb2;
  bool m = mask[n] != 0;
  float x0 = m ? 1e6f : x_ca[n * 3 + 0];
  float x1 = m ? 1e6f : x_ca[n * 3 + 1];
  float x2 = m ? 1e6f : x_ca[n * 3 + 2];
  xm[n * 3 + 0] = x0; xm[n * 3 + 1] = x1; xm[n * 3 + 2] = x2;
  sq[n] = x0 * x0 + x1 * x1 + x2 * x2;
}

// =====================================================================
// Per-node K-nearest-neighbors (block per node, d2 in LDS, iterated argmin)
// =====================================================================
__global__ __launch_bounds__(256) void topk_kernel(
    const float* __restrict__ xm, const float* __restrict__ sq,
    const int* __restrict__ batch, int* __restrict__ neigh, int* __restrict__ centers)
{
  __shared__ float d2s[NN];
  __shared__ float rv[256];
  __shared__ int   ri[256];
  int i = blockIdx.x;
  float xi0 = xm[i * 3], xi1 = xm[i * 3 + 1], xi2 = xm[i * 3 + 2];
  float sqi = sq[i];
  int bi = batch[i];
  for (int j = threadIdx.x; j < NN; j += 256) {
    float d2 = sqi + sq[j] -
               2.f * (xi0 * xm[j * 3] + xi1 * xm[j * 3 + 1] + xi2 * xm[j * 3 + 2]);
    if (batch[j] != bi || j == i) d2 = __builtin_inff();
    d2s[j] = d2;
  }
  __syncthreads();
  for (int it = 0; it < KNB; ++it) {
    float best = __builtin_inff();
    int bidx = 0x7fffffff;
    for (int j = threadIdx.x; j < NN; j += 256) {
      float v = d2s[j];
      if (v < best || (v == best && j < bidx)) { best = v; bidx = j; }
    }
    rv[threadIdx.x] = best; ri[threadIdx.x] = bidx;
    __syncthreads();
    for (int s = 128; s; s >>= 1) {
      if (threadIdx.x < s) {
        float ov = rv[threadIdx.x + s]; int oi = ri[threadIdx.x + s];
        if (ov < rv[threadIdx.x] || (ov == rv[threadIdx.x] && oi < ri[threadIdx.x])) {
          rv[threadIdx.x] = ov; ri[threadIdx.x] = oi;
        }
      }
      __syncthreads();
    }
    if (threadIdx.x == 0) {
      int w = ri[0];
      neigh[i * KNB + it]   = w;
      centers[i * KNB + it] = i;
      d2s[w] = __builtin_inff();
    }
    __syncthreads();
  }
}

// =====================================================================
// Edge features: 25 pairwise distances -> 16 RBF each, + 16 sinusoidal pos
// =====================================================================
__global__ __launch_bounds__(128) void edge_feat_kernel(
    const float* __restrict__ bb5, const int* __restrict__ neigh,
    float* __restrict__ ef)
{
  int e = blockIdx.x * blockDim.x + threadIdx.x;
  if (e >= EE) return;
  int c  = e / KNB;
  int nb = neigh[e];
  float pc[15], pn[15];
  for (int i = 0; i < 15; ++i) { pc[i] = bb5[(size_t)c * 15 + i]; pn[i] = bb5[(size_t)nb * 15 + i]; }
  float* o = ef + (size_t)e * EFD;
  for (int a = 0; a < 5; ++a)
    for (int b = 0; b < 5; ++b) {
      float dx = pc[a * 3 + 0] - pn[b * 3 + 0] + 1e-8f;
      float dy = pc[a * 3 + 1] - pn[b * 3 + 1] + 1e-8f;
      float dz = pc[a * 3 + 2] - pn[b * 3 + 2] + 1e-8f;
      float d = sqrtf(dx * dx + dy * dy + dz * dz);
      for (int r = 0; r < 16; ++r) {
        float mu = 2.f + (float)r * (20.f / 15.f);
        float t = (d - mu) * 0.8f;
        o[(a * 5 + b) * 16 + r] = __expf(-t * t);
      }
    }
  float rel = (float)(nb - c);
  for (int k = 0; k < 8; ++k) {
    float fr = __expf(-(float)k * 1.1512925465f);
    float ang = rel * fr;
    o[400 + k] = __cosf(ang);
    o[408 + k] = __sinf(ang);
  }
}

// =====================================================================
// IPMP geometry: g = R@pts + t  ;  q = R[c]^T (g[nb] - t[c]), dist = |q|
// =====================================================================
__global__ __launch_bounds__(256) void frames_g_kernel(
    const float* __restrict__ R, const float* __restrict__ t,
    const float* __restrict__ pts, float* __restrict__ g)
{
  int idx = blockIdx.x * blockDim.x + threadIdx.x;
  if (idx >= NN * 4) return;
  int n = idx >> 2, p = idx & 3;
  const float* Rn = R + (size_t)n * 9;
  float p0 = pts[n * 12 + p * 3 + 0];
  float p1 = pts[n * 12 + p * 3 + 1];
  float p2 = pts[n * 12 + p * 3 + 2];
  for (int i = 0; i < 3; ++i)
    g[n * 12 + p * 3 + i] =
        Rn[i * 3 + 0] * p0 + Rn[i * 3 + 1] * p1 + Rn[i * 3 + 2] * p2 + t[n * 3 + i];
}

__global__ __launch_bounds__(256) void qdist_kernel(
    const float* __restrict__ R, const float* __restrict__ t,
    const float* __restrict__ g, const int* __restrict__ neigh,
    float* __restrict__ qb, float* __restrict__ db)
{
  int idx = blockIdx.x * blockDim.x + threadIdx.x;
  if (idx >= EE * 4) return;
  int e = idx >> 2, p = idx & 3;
  int c  = e / KNB;
  int nb = neigh[e];
  const float* Rc = R + (size_t)c * 9;
  float v0 = g[nb * 12 + p * 3 + 0] - t[c * 3 + 0];
  float v1 = g[nb * 12 + p * 3 + 1] - t[c * 3 + 1];
  float v2 = g[nb * 12 + p * 3 + 2] - t[c * 3 + 2];
  float q0 = Rc[0] * v0 + Rc[3] * v1 + Rc[6] * v2;   // R^T v
  float q1 = Rc[1] * v0 + Rc[4] * v1 + Rc[7] * v2;
  float q2 = Rc[2] * v0 + Rc[5] * v1 + Rc[8] * v2;
  qb[(size_t)e * 12 + p * 3 + 0] = q0;
  qb[(size_t)e * 12 + p * 3 + 1] = q1;
  qb[(size_t)e * 12 + p * 3 + 2] = q2;
  db[(size_t)e * 4 + p] = sqrtf(q0 * q0 + q1 * q1 + q2 * q2);
}

// segment mean over the K contiguous edges of each node
__global__ __launch_bounds__(256) void agg_kernel(
    const float* __restrict__ h2, float* __restrict__ agg)
{
  int idx = blockIdx.x * blockDim.x + threadIdx.x;
  if (idx >= NN * CH_) return;
  int n = idx >> 7, c = idx & 127;
  float s = 0.f;
  for (int k = 0; k < KNB; ++k) s += h2[((size_t)n * KNB + k) * CH_ + c];
  agg[idx] = s * (1.f / (float)KNB);
}

__global__ __launch_bounds__(256) void copy_kernel(
    const float* __restrict__ src, float* __restrict__ dst, size_t n)
{
  size_t i = (size_t)blockIdx.x * blockDim.x + threadIdx.x;
  if (i < n) dst[i] = src[i];
}

// torsion normalize: u <- u / |u + 1e-8|
__global__ __launch_bounds__(256) void unorm_kernel(float* __restrict__ u)
{
  int idx = blockIdx.x * blockDim.x + threadIdx.x;
  if (idx >= NN * 81) return;
  int n = idx / 81, p = idx % 81;
  float* pp = u + (size_t)n * 162 + p * 2;
  float a = pp[0], b = pp[1];
  float a2 = a + 1e-8f, b2 = b + 1e-8f;
  float nr = sqrtf(a2 * a2 + b2 * b2);
  pp[0] = a / nr; pp[1] = b / nr;
}

// =====================================================================
// All-atom reconstruction
// =====================================================================
__device__ float g_LIT[20 * 14 * 3];   // LIT_POS stand-in (module RNG constant)
__device__ float g_GT[20 * 4 * 3];     // GROUP_TRANS stand-in

__device__ __forceinline__ float hash_normal(unsigned i)
{
  float acc = 0.f;
  for (int k = 0; k < 4; ++k) {
    unsigned x = i * 4u + (unsigned)k + 0x9e3779b9u;
    x ^= x >> 16; x *= 0x7feb352du; x ^= x >> 15; x *= 0x846ca68bu; x ^= x >> 16;
    acc += (float)(x >> 8) * (1.f / 16777216.f);
  }
  return (acc - 2.f) * 1.7320508f;
}

__global__ void init_consts_kernel()
{
  int i = blockIdx.x * blockDim.x + threadIdx.x;
  if (i < 20 * 14 * 3) g_LIT[i] = hash_normal(i);
  if (i < 20 * 4 * 3)  g_GT[i]  = hash_normal(i + 100000u);
}

__device__ const int c_SC[20] = {1,7,4,4,2,5,5,0,6,4,4,5,4,7,3,2,3,10,8,3};

__device__ __forceinline__ void rot_rx(float* Rm, float s, float c)
{
  for (int i = 0; i < 3; ++i) {
    float b = Rm[i * 3 + 1], d = Rm[i * 3 + 2];
    Rm[i * 3 + 1] = c * b + s * d;
    Rm[i * 3 + 2] = -s * b + c * d;
  }
}

__global__ __launch_bounds__(256) void atom_kernel(
    const float* __restrict__ R, const float* __restrict__ t,
    const float* __restrict__ u, float* __restrict__ out)
{
  int idx = blockIdx.x * blockDim.x + threadIdx.x;
  if (idx >= NN * 91) return;
  int n = idx / 91, j = idx % 91;
  int aa = 0, atom = j;
  if (j >= 4) {
    int pos = 4;
    for (int a = 0; a < 20; ++a) {
      int cnt = c_SC[a];
      if (j < pos + cnt) { aa = a; atom = 4 + (j - pos); break; }
      pos += cnt;
    }
  }
  int f;
  if (atom < 3) f = 0;
  else if (atom == 3) f = 1;
  else { int gi = (atom - 4) / 3; if (gi > 3) gi = 3; f = 2 + gi; }

  float Rg[9], tg[3], tn[3];
  for (int k = 0; k < 9; ++k) Rg[k] = R[(size_t)n * 9 + k];
  for (int k = 0; k < 3; ++k) { tn[k] = t[(size_t)n * 3 + k]; tg[k] = tn[k]; }

  if (f == 1) {
    rot_rx(Rg, u[(size_t)n * 162 + 0], u[(size_t)n * 162 + 1]);     // psi
  } else if (f >= 2) {
    for (int g2 = 0; g2 <= f - 2; ++g2) {
      const float* gt = &g_GT[(aa * 4 + g2) * 3];
      for (int i = 0; i < 3; ++i)
        tg[i] += Rg[i * 3 + 0] * gt[0] + Rg[i * 3 + 1] * gt[1] + Rg[i * 3 + 2] * gt[2];
      int p = 1 + aa * 4 + g2;
      rot_rx(Rg, u[(size_t)n * 162 + p * 2], u[(size_t)n * 162 + p * 2 + 1]);
    }
  }
  const float* lp = &g_LIT[(aa * 14 + atom) * 3];
  for (int i = 0; i < 3; ++i)
    out[(size_t)idx * 3 + i] =
        Rg[i * 3 + 0] * lp[0] + Rg[i * 3 + 1] * lp[1] + Rg[i * 3 + 2] * lp[2]
        + tg[i] - tn[i];
}

// =====================================================================
// Host orchestration
// =====================================================================
struct LayerP {
  const float *et_b1, *et_fin_b, *et_fin_w, *et_init_b, *et_init_w, *et_l2_b,
              *et_l2_w, *et_ln_b, *et_ln_g, *et_w_c, *et_w_e, *et_w_n;
  const float *ip_b1, *ip_l2_b, *ip_l2_w, *ip_out_b, *ip_out_w, *ip_pts_b,
              *ip_pts_w, *ip_w_d, *ip_w_q, *ip_w_si, *ip_w_sj, *ip_w_z;
  const float *ln_b, *ln_g;
  const float *nt_l1_b, *nt_l1_w, *nt_l2_b, *nt_l2_w, *nt_l3_b, *nt_l3_w,
              *nt_ln_b, *nt_ln_g;
};
struct LayerW {
  const __bf16 *et_fin, *et_init, *et_l2, *et_w_c, *et_w_e, *et_w_n;
  const __bf16 *ip_l2, *ip_out, *ip_pts, *ip_w_d, *ip_w_q, *ip_w_si, *ip_w_sj, *ip_w_z;
  const __bf16 *nt_l1, *nt_l2, *nt_l3;
};

extern "C" void kernel_launch(void* const* d_in, const int* in_sizes, int n_in,
                              void* d_out, int out_size, void* d_ws, size_t ws_size,
                              hipStream_t stream)
{
  (void)in_sizes; (void)n_in; (void)out_size; (void)ws_size;

  // ---- tensor inputs (setup_inputs dict order) ----
  const float* x_ca = (const float*)d_in[0];
  const float* bb   = (const float*)d_in[1];
  const float* R    = (const float*)d_in[2];
  const float* t    = (const float*)d_in[3];
  const float* latent = (const float*)d_in[4];
  const unsigned char* x_mask = (const unsigned char*)d_in[5];
  const int* batch  = (const int*)d_in[6];

  // ---- params, pytree (alphabetical-key) flatten order ----
  int pi = 7;
  auto P = [&]() -> const float* { return (const float*)d_in[pi++]; };
  const float *ee_l1_b = P(), *ee_l1_w = P(), *ee_l2_b = P(), *ee_l2_w = P(),
              *ee_l3_b = P(), *ee_l3_w = P(), *ee_ln_b = P(), *ee_ln_g = P();
  LayerP L[4];
  for (int i = 0; i < 4; ++i) {
    L[i].et_b1 = P(); L[i].et_fin_b = P(); L[i].et_fin_w = P();
    L[i].et_init_b = P(); L[i].et_init_w = P();
    L[i].et_l2_b = P(); L[i].et_l2_w = P();
    L[i].et_ln_b = P(); L[i].et_ln_g = P();
    L[i].et_w_c = P(); L[i].et_w_e = P(); L[i].et_w_n = P();
    L[i].ip_b1 = P(); L[i].ip_l2_b = P(); L[i].ip_l2_w = P();
    L[i].ip_out_b = P(); L[i].ip_out_w = P();
    L[i].ip_pts_b = P(); L[i].ip_pts_w = P();
    L[i].ip_w_d = P(); L[i].ip_w_q = P();
    L[i].ip_w_si = P(); L[i].ip_w_sj = P(); L[i].ip_w_z = P();
    L[i].ln_b = P(); L[i].ln_g = P();
    L[i].nt_l1_b = P(); L[i].nt_l1_w = P();
    L[i].nt_l2_b = P(); L[i].nt_l2_w = P();
    L[i].nt_l3_b = P(); L[i].nt_l3_w = P();
    L[i].nt_ln_b = P(); L[i].nt_ln_g = P();
  }
  const float *seq_b = P(), *seq_w = P(), *tor_b = P(), *tor_w = P();

  // ---- workspace carve ----
  char* wsb = (char*)d_ws;
  size_t off = 0;
  auto allocf = [&](size_t n) -> float* {
    float* p = (float*)(wsb + off);
    off = (off + n * 4 + 255) & ~(size_t)255;
    return p;
  };
  auto alloci = [&](size_t n) -> int* {
    int* p = (int*)(wsb + off);
    off = (off + n * 4 + 255) & ~(size_t)255;
    return p;
  };
  auto allocbf = [&](size_t n) -> __bf16* {
    __bf16* p = (__bf16*)(wsb + off);
    off = (off + n * 2 + 255) & ~(size_t)255;
    return p;
  };
  float* bb5  = allocf((size_t)NN * 15);
  float* xm   = allocf((size_t)NN * 3);
  float* sq   = allocf((size_t)NN);
  int*   neigh   = alloci((size_t)EE);
  int*   centers = alloci((size_t)EE);
  float* buf0 = allocf((size_t)EE * EFD);   // ef / ipmp-h / et-h
  float* buf1 = allocf((size_t)EE * HET_);  // embed t1 / ipmp-h2 / et-h2
  float* zbuf = allocf((size_t)EE * CZ_);
  float* sbuf = allocf((size_t)NN * CS_);
  float* ptsb = allocf((size_t)NN * 12);
  float* gb   = allocf((size_t)NN * 12);
  float* qb   = allocf((size_t)EE * 12);
  float* db   = allocf((size_t)EE * 4);
  float* aggb = allocf((size_t)NN * CH_);
  float* updb = allocf((size_t)NN * CS_);
  float* neb  = allocf((size_t)NN * CZ_);
  float* ubuf = allocf((size_t)NN * 162);

  // ---- one-time bf16 weight transposes (W[K][N] f32 -> Wt[N][K] bf16) ----
  auto prepW = [&](const float* W, int Kd, int Nd) -> const __bf16* {
    __bf16* p = allocbf((size_t)Kd * Nd);
    int tot = Kd * Nd;
    wtrans_kernel<<<(tot + 255) / 256, 256, 0, stream>>>(W, p, Kd, Nd);
    return p;
  };
  const __bf16* ee_l1_t = prepW(ee_l1_w, EFD, 2 * CZ_);
  const __bf16* ee_l2_t = prepW(ee_l2_w, 2 * CZ_, 2 * CZ_);
  const __bf16* ee_l3_t = prepW(ee_l3_w, 2 * CZ_, CZ_);
  LayerW LW[4];
  for (int i = 0; i < 4; ++i) {
    LW[i].et_fin  = prepW(L[i].et_fin_w, HET_, CZ_);
    LW[i].et_init = prepW(L[i].et_init_w, CS_, CS_ / 2);
    LW[i].et_l2   = prepW(L[i].et_l2_w, HET_, HET_);
    LW[i].et_w_c  = prepW(L[i].et_w_c, CS_ / 2, HET_);
    LW[i].et_w_e  = prepW(L[i].et_w_e, CZ_, HET_);
    LW[i].et_w_n  = prepW(L[i].et_w_n, CS_ / 2, HET_);
    LW[i].ip_l2   = prepW(L[i].ip_l2_w, CH_, CH_);
    LW[i].ip_out  = prepW(L[i].ip_out_w, CH_, CS_);
    LW[i].ip_pts  = prepW(L[i].ip_pts_w, CS_, 12);
    LW[i].ip_w_d  = prepW(L[i].ip_w_d, 4, CH_);
    LW[i].ip_w_q  = prepW(L[i].ip_w_q, 12, CH_);
    LW[i].ip_w_si = prepW(L[i].ip_w_si, CS_, CH_);
    LW[i].ip_w_sj = prepW(L[i].ip_w_sj, CS_, CH_);
    LW[i].ip_w_z  = prepW(L[i].ip_w_z, CZ_, CH_);
    LW[i].nt_l1   = prepW(L[i].nt_l1_w, CS_, CS_);
    LW[i].nt_l2   = prepW(L[i].nt_l2_w, CS_, CS_);
    LW[i].nt_l3   = prepW(L[i].nt_l3_w, CS_, CS_);
  }
  const __bf16* tor_t = prepW(tor_w, CS_, 162);
  const __bf16* seq_t = prepW(seq_w, CS_, 20);

  auto gemm = [&](const float* A, const int* rowIdx, const __bf16* Wt,
                  const float* bias, float* C, int M, int Nd, int Kd,
                  int beta, int act) {
    dim3 g((M + 127) / 128, (Nd + 127) / 128);
    bool bounds = (M & 127) || (Nd & 127) || (Kd & 31);
    if (bounds)
      gemm_bf16_wmma<true><<<g, dim3(256), 0, stream>>>(A, rowIdx, Wt, bias, C, M, Nd, Kd, beta, act);
    else
      gemm_bf16_wmma<false><<<g, dim3(256), 0, stream>>>(A, rowIdx, Wt, bias, C, M, Nd, Kd, beta, act);
  };
  auto ln = [&](const float* x, const float* upd, const float* gg, const float* bbp,
                const unsigned char* mk, int mode, float* outp, int M, int D) {
    ln_kernel<<<(M + 7) / 8, 256, 0, stream>>>(x, upd, gg, bbp, mk, mode, outp, M, D);
  };

  float* dout = (float*)d_out;

  // ---- constants + geometry + graph ----
  init_consts_kernel<<<(20 * 14 * 3 + 63) / 64, 64, 0, stream>>>();
  prep_kernel<<<(NN + 255) / 256, 256, 0, stream>>>(x_ca, bb, x_mask, bb5, xm, sq);
  topk_kernel<<<NN, 256, 0, stream>>>(xm, sq, batch, neigh, centers);
  edge_feat_kernel<<<(EE + 127) / 128, 128, 0, stream>>>(bb5, neigh, buf0);

  // ---- edge embedding ----
  gemm(buf0, nullptr, ee_l1_t, ee_l1_b, buf1, EE, 2 * CZ_, EFD, 0, 1);
  gemm(buf1, nullptr, ee_l2_t, ee_l2_b, buf0, EE, 2 * CZ_, 2 * CZ_, 0, 1);
  gemm(buf0, nullptr, ee_l3_t, ee_l3_b, zbuf, EE, CZ_, 2 * CZ_, 0, 0);
  ln(zbuf, nullptr, ee_ln_g, ee_ln_b, nullptr, 0, zbuf, EE, CZ_);

  // s = latent
  copy_kernel<<<((size_t)NN * CS_ + 255) / 256, 256, 0, stream>>>(latent, sbuf, (size_t)NN * CS_);

  for (int li = 0; li < 4; ++li) {
    const LayerP& p = L[li];
    const LayerW& w = LW[li];
    // ---- IPMP ----
    gemm(sbuf, nullptr, w.ip_pts, p.ip_pts_b, ptsb, NN, 12, CS_, 0, 0);
    frames_g_kernel<<<(NN * 4 + 255) / 256, 256, 0, stream>>>(R, t, ptsb, gb);
    qdist_kernel<<<(EE * 4 + 255) / 256, 256, 0, stream>>>(R, t, gb, neigh, qb, db);
    gemm(sbuf, centers, w.ip_w_si, nullptr, buf0, EE, CH_, CS_, 0, 0);
    gemm(sbuf, neigh,   w.ip_w_sj, nullptr, buf0, EE, CH_, CS_, 1, 0);
    gemm(zbuf, nullptr, w.ip_w_z,  nullptr, buf0, EE, CH_, CZ_, 1, 0);
    gemm(qb,   nullptr, w.ip_w_q,  nullptr, buf0, EE, CH_, 12, 1, 0);
    gemm(db,   nullptr, w.ip_w_d,  p.ip_b1, buf0, EE, CH_, 4, 1, 1);
    gemm(buf0, nullptr, w.ip_l2, p.ip_l2_b, buf1, EE, CH_, CH_, 0, 1);
    agg_kernel<<<(NN * CH_ + 255) / 256, 256, 0, stream>>>(buf1, aggb);
    gemm(aggb, nullptr, w.ip_out, p.ip_out_b, updb, NN, CS_, CH_, 0, 0);
    ln(sbuf, updb, p.ln_g, p.ln_b, x_mask, 1, sbuf, NN, CS_);

    // ---- node transition ----
    gemm(sbuf, nullptr, w.nt_l1, p.nt_l1_b, buf0, NN, CS_, CS_, 0, 1);
    gemm(buf0, nullptr, w.nt_l2, p.nt_l2_b, buf1, NN, CS_, CS_, 0, 1);
    gemm(buf1, nullptr, w.nt_l3, p.nt_l3_b, updb, NN, CS_, CS_, 0, 0);
    ln(sbuf, updb, p.nt_ln_g, p.nt_ln_b, x_mask, 2, sbuf, NN, CS_);

    // ---- edge transition ----
    gemm(sbuf, nullptr, w.et_init, p.et_init_b, neb, NN, CS_ / 2, CS_, 0, 0);
    gemm(zbuf, nullptr, w.et_w_e, nullptr, buf0, EE, HET_, CZ_, 0, 0);
    gemm(neb,  centers, w.et_w_c, nullptr, buf0, EE, HET_, CS_ / 2, 1, 0);
    gemm(neb,  neigh,   w.et_w_n, p.et_b1, buf0, EE, HET_, CS_ / 2, 1, 1);
    gemm(buf0, nullptr, w.et_l2, p.et_l2_b, buf1, EE, HET_, HET_, 0, 1);
    gemm(buf1, nullptr, w.et_fin, p.et_fin_b, zbuf, EE, CZ_, HET_, 0, 0);
    ln(zbuf, nullptr, p.et_ln_g, p.et_ln_b, nullptr, 0, zbuf, EE, CZ_);
  }

  // ---- heads ----
  gemm(sbuf, nullptr, tor_t, tor_b, ubuf, NN, 162, CS_, 0, 0);
  unorm_kernel<<<(NN * 81 + 255) / 256, 256, 0, stream>>>(ubuf);
  gemm(sbuf, nullptr, seq_t, seq_b, dout + (size_t)NN * 91 * 3, NN, 20, CS_, 0, 0);
  atom_kernel<<<(NN * 91 + 255) / 256, 256, 0, stream>>>(R, t, ubuf, dout);
}
